// LSTMAgent_85890755985408
// MI455X (gfx1250) — matrix operand order repaired and checked by reading, not compiled, not run on hardware
//
#include <hip/hip_runtime.h>
#include <hip/hip_bf16.h>

// ---------------------------------------------------------------------------
// MI455X (gfx1250) implementation of the conv+LSTM agent forward pass.
// Everything matrix-shaped runs through V_WMMA_F32_16X16X32_F16 (wave32).
// Each wave owns a 16 x (16*NT) output strip so one A-fragment feeds NT WMMAs.
// Workspace requirement: ~78 MB (bump-allocated with region aliasing).
// ---------------------------------------------------------------------------

typedef _Float16 h16;
typedef __attribute__((ext_vector_type(16))) _Float16 v16h;
typedef __attribute__((ext_vector_type(8)))  _Float16 v8h;
typedef __attribute__((ext_vector_type(8)))  float    v8f;

union V16U { v16h v; v8h h[2]; };

#define LOG2PI_F 1.8378770664093454f

enum { ACT_NONE = 0, ACT_RELU = 1, ACT_TANH = 2 };

__device__ __forceinline__ float sigmoidf_(float x) { return 1.0f / (1.0f + __expf(-x)); }

// A-matrix fragment (f16 16x32): lane holds row M = lane%16.
// halves e=0..15 -> K = k0 + 16*(lane/16) + e (matches frag_colK)
// => one contiguous 16-half (32B) region.
__device__ __forceinline__ v16h frag_rowK(const h16* base, int stride, int row, int k0, int lane) {
  const int hs = lane >> 4;
  const h16* p = base + (size_t)row * stride + k0 + 16 * hs;
  V16U r;
  r.h[0] = *(const v8h*)(p);
  r.h[1] = *(const v8h*)(p + 8);
  return r.v;
}

// B-matrix fragment (f16 32x16) from weights stored [N][K] (i.e. w[cout][k]):
// lane holds column n = lane%16; halves e=0..15 -> K = k0 + 16*(lane/16) + e
// => one contiguous 16-half (32B) region.
__device__ __forceinline__ v16h frag_colK(const h16* base, int stride, int nrow, int k0, int lane) {
  const int hs = lane >> 4;
  const h16* p = base + (size_t)nrow * stride + k0 + 16 * hs;
  V16U r;
  r.h[0] = *(const v8h*)(p);
  r.h[1] = *(const v8h*)(p + 8);
  return r.v;
}

// ---------------------------------------------------------------------------
// Generic f16 WMMA GEMM:  C[M][N] = act( A[M][lda] * B(N rows of K)^T + bias )
//   A: f16 row-major (lda halves per row), B: f16 [N][K] ("w[cout][k]")
//   NT:    n-tiles per wave (strip); one A-fragment feeds NT WMMAs
//   CIN:   accumulate on top of f32 C-input (folds gates_x into LSTM step)
//   MASKA: per-row mask (1 - done[row]) applied to A (LSTM reset)
// M, N multiples of 16; N/16 multiple of NT; K multiple of 32.
// ---------------------------------------------------------------------------
template<int NT, int ACT, bool OUT16, bool CIN, bool MASKA>
__global__ void gemm_wmma_f16(const h16* __restrict__ A, int lda,
                              const h16* __restrict__ B, int ldb,
                              const float* __restrict__ bias,
                              const float* __restrict__ Cin, int ldcin,
                              void* __restrict__ Cout, int ldc,
                              const int* __restrict__ doneRow,
                              int M, int N, int K)
{
  const int lane     = threadIdx.x & 31;
  const int wave     = threadIdx.x >> 5;
  const int strips_n = (N >> 4) / NT;
  const int total    = (M >> 4) * strips_n;
  const int tile     = blockIdx.x * (blockDim.x >> 5) + wave;
  if (tile >= total) return;                 // whole-wave exit: EXEC stays all-ones
  const int tm  = tile / strips_n;
  const int sn  = tile - tm * strips_n;
  const int nl  = lane & 15, hs = lane >> 4;
  const int rowA = tm * 16 + nl;
  const int nb0  = sn * NT * 16;             // first output column block of strip

  h16 msc = (h16)1.0f;
  if (MASKA) msc = (h16)(1.0f - (float)doneRow[rowA]);

  v8f acc[NT];
  if (CIN) {
#pragma unroll
    for (int t = 0; t < NT; t++)
#pragma unroll
      for (int i = 0; i < 8; i++)
        acc[t][i] = Cin[(size_t)(tm * 16 + i + 8 * hs) * ldcin + nb0 + t * 16 + nl];
  } else {
#pragma unroll
    for (int t = 0; t < NT; t++)
#pragma unroll
      for (int i = 0; i < 8; i++) acc[t][i] = 0.0f;
  }

  for (int k0 = 0; k0 < K; k0 += 32) {
    if (k0 + 32 < K) {
      __builtin_prefetch(A + (size_t)rowA * lda + k0 + 32, 0, 1);            // global_prefetch_b8
      __builtin_prefetch(B + (size_t)(nb0 + nl) * ldb + k0 + 32, 0, 1);
    }
    v16h a = frag_rowK(A, lda, rowA, k0, lane);
    if (MASKA) {
#pragma unroll
      for (int i = 0; i < 16; i++) a[i] = a[i] * msc;
    }
#pragma unroll
    for (int t = 0; t < NT; t++) {
      v16h b = frag_colK(B, ldb, nb0 + t * 16 + nl, k0, lane);
      acc[t] = __builtin_amdgcn_wmma_f32_16x16x32_f16(false, a, false, b, (short)0, acc[t], false, false);
    }
  }

#pragma unroll
  for (int t = 0; t < NT; t++) {
    const float bv = bias ? bias[nb0 + t * 16 + nl] : 0.0f;
#pragma unroll
    for (int i = 0; i < 8; i++) {
      float v = acc[t][i] + bv;
      if (ACT == ACT_RELU) v = v > 0.0f ? v : 0.0f;
      if (ACT == ACT_TANH) v = tanhf(v);
      const int r = tm * 16 + i + 8 * hs, c = nb0 + t * 16 + nl;
      if (OUT16) ((h16*)Cout)[(size_t)r * ldc + c]  = (h16)v;
      else       ((float*)Cout)[(size_t)r * ldc + c] = v;
    }
  }
}

// ---------------------------------------------------------------------------
// Implicit-GEMM conv + bias + relu via WMMA. One wave per 16-row M tile,
// computing ALL COUT columns (NT = COUT/16 WMMAs per K-step reuse one
// LDS-staged A tile). Row decodes are hoisted: per-row global base offsets are
// computed once; each K-step adds a single lane-constant im2col offset.
//   GEMM view: M = N*HOUT*WOUT, N = COUT, K = CIN*KH*KW.
// FROMX folds the reference's (0,3,2,1) transpose of the fp32 NHWC input
// into the gather.
// ---------------------------------------------------------------------------
template<int CIN, int KH, int KW, int S, int HIN, int WIN,
         int COUT, int HOUT, int WOUT, bool FROMX>
__global__ void conv_wmma_f16(const void* __restrict__ src,
                              const h16* __restrict__ W,
                              const float* __restrict__ bias,
                              h16* __restrict__ dst)
{
  constexpr int K  = CIN * KH * KW;
  constexpr int NT = COUT / 16;
  __shared__ __attribute__((aligned(32))) h16 atile[16 * 32];
  const int lane = threadIdx.x;
  const int tm   = blockIdx.x;
  const int nl = lane & 15, hs = lane >> 4;

  // Per-row global base offsets (element units), row decode done once.
  int rowbase[16];
#pragma unroll
  for (int r = 0; r < 16; r++) {
    const int gm = tm * 16 + r;
    const int ow = gm % WOUT;
    const int t1 = gm / WOUT;
    const int oh = t1 % HOUT;
    const int n  = t1 / HOUT;
    rowbase[r] = FROMX ? ((n * WIN + ow * S) * HIN + oh * S) * CIN
                       : ((n * CIN) * HIN + oh * S) * WIN + ow * S;
  }

  v8f acc[NT];
#pragma unroll
  for (int t = 0; t < NT; t++)
#pragma unroll
    for (int i = 0; i < 8; i++) acc[t][i] = 0.0f;

  for (int k0 = 0; k0 < K; k0 += 32) {
    const int kg = k0 + lane;                 // this lane's fixed k column
    const int kw = kg % KW;
    const int kh = (kg / KW) % KH;
    const int ci = kg / (KW * KH);
    const int koff = FROMX ? (kw * HIN + kh) * CIN + ci
                           : ci * HIN * WIN + kh * WIN + kw;
#pragma unroll
    for (int r = 0; r < 16; r++) {
      float v = FROMX ? ((const float*)src)[rowbase[r] + koff]
                      : (float)((const h16*)src)[rowbase[r] + koff];
      atile[r * 32 + lane] = (h16)v;
    }
    __syncthreads();
    v16h a = frag_rowK(atile, 32, nl, 0, lane);
#pragma unroll
    for (int t = 0; t < NT; t++) {
      v16h b = frag_colK(W, K, t * 16 + nl, k0, lane);
      acc[t] = __builtin_amdgcn_wmma_f32_16x16x32_f16(false, a, false, b, (short)0, acc[t], false, false);
    }
    __syncthreads();
  }

  // Epilogue: precompute per-i output bases, then store NT tiles.
  int obase[8];
#pragma unroll
  for (int i = 0; i < 8; i++) {
    const int gm = tm * 16 + i + 8 * hs;
    const int ow = gm % WOUT;
    const int t1 = gm / WOUT;
    const int oh = t1 % HOUT;
    const int n  = t1 / HOUT;
    obase[i] = (n * COUT) * (HOUT * WOUT) + oh * WOUT + ow;
  }
#pragma unroll
  for (int t = 0; t < NT; t++) {
    const int cc = t * 16 + nl;
    const float bv = bias[cc];
#pragma unroll
    for (int i = 0; i < 8; i++) {
      float v = acc[t][i] + bv;
      v = v > 0.0f ? v : 0.0f;               // relu
      dst[(size_t)obase[i] + (size_t)cc * (HOUT * WOUT)] = (h16)v;
    }
  }
}

// ------------------------------ small kernels ------------------------------

__global__ void k_pack_pad_f16(const float* __restrict__ src, h16* __restrict__ dst,
                               int sR, int sC, int dR, int dC) {
  int i = blockIdx.x * blockDim.x + threadIdx.x;
  if (i >= dR * dC) return;
  int r = i / dC, c = i - r * dC;
  dst[i] = (r < sR && c < sC) ? (h16)src[r * sC + c] : (h16)0.0f;
}

__global__ void k_pad_f32(const float* __restrict__ src, float* __restrict__ dst, int sN, int dN) {
  int i = blockIdx.x * blockDim.x + threadIdx.x;
  if (i >= dN) return;
  dst[i] = (i < sN) ? src[i] : 0.0f;
}

__global__ void k_add_f32(const float* a, const float* b, float* o, int n) {
  int i = blockIdx.x * blockDim.x + threadIdx.x;
  if (i < n) o[i] = a[i] + b[i];
}

__global__ void k_seq_tail(const float* __restrict__ la, const float* __restrict__ lr,
                           h16* __restrict__ aseq) {
  int n = blockIdx.x * blockDim.x + threadIdx.x;
  if (n >= 2048) return;
  h16* row = aseq + (size_t)n * 544;
  row[512] = (h16)la[n * 2 + 0];
  row[513] = (h16)la[n * 2 + 1];
  row[514] = (h16)lr[n];
  for (int k = 515; k < 544; k++) row[k] = (h16)0.0f;
}

__global__ void k_init_state(const float* h0, const float* c0, h16* hq, float* c) {
  int i = blockIdx.x * blockDim.x + threadIdx.x;
  if (i >= 128 * 128) return;
  hq[i] = (h16)h0[i];
  c[i]  = c0[i];
}

__global__ void k_lstm_elem(const float* __restrict__ g, const int* __restrict__ done_t,
                            float* __restrict__ c, h16* __restrict__ hq,
                            h16* __restrict__ hiddenT) {
  int i = blockIdx.x * blockDim.x + threadIdx.x;
  if (i >= 128 * 128) return;
  int b = i >> 7, j = i & 127;
  const float* gr = g + (size_t)b * 512;
  float m  = 1.0f - (float)done_t[b];
  float ig = sigmoidf_(gr[j]);
  float fg = sigmoidf_(gr[128 + j]);
  float gg = tanhf(gr[256 + j]);
  float og = sigmoidf_(gr[384 + j]);
  float cs = fg * (c[i] * m) + ig * gg;
  float hs = og * tanhf(cs);
  c[i] = cs;
  hq[i] = (h16)hs;
  hiddenT[i] = (h16)hs;
}

__global__ void k_finalize(const float* __restrict__ mu3, const float* __restrict__ v3,
                           const float* __restrict__ action, const float* __restrict__ logstd,
                           float* __restrict__ out) {
  int n = blockIdx.x * blockDim.x + threadIdx.x;
  if (n >= 2048) return;
  float ls0 = logstd[0], ls1 = logstd[1];
  float s0 = __expf(ls0), s1 = __expf(ls1);
  float m0 = mu3[n * 16 + 0], m1 = mu3[n * 16 + 1];
  float a0 = action[n * 2 + 0], a1 = action[n * 2 + 1];
  float z0 = (a0 - m0) / s0, z1 = (a1 - m1) / s1;
  out[n * 3 + 0] = -0.5f * (z0 * z0 + z1 * z1) - (ls0 + ls1) - LOG2PI_F;
  out[n * 3 + 1] = (1.0f + LOG2PI_F) + ls0 + ls1;
  out[n * 3 + 2] = v3[n * 16 + 0];
}

// ------------------------------ host launcher ------------------------------

extern "C" void kernel_launch(void* const* d_in, const int* in_sizes, int n_in,
                              void* d_out, int out_size, void* d_ws, size_t ws_size,
                              hipStream_t stream) {
  (void)in_sizes; (void)n_in; (void)out_size; (void)ws_size;

  const float* x    = (const float*)d_in[0];
  const int*   done = (const int*)  d_in[1];
  const float* la   = (const float*)d_in[2];
  const float* lr   = (const float*)d_in[3];
  const float* h0   = (const float*)d_in[4];
  const float* c0   = (const float*)d_in[5];
  const float* actn = (const float*)d_in[6];
  const float* c1w = (const float*)d_in[7];  const float* c1b = (const float*)d_in[8];
  const float* c2w = (const float*)d_in[9];  const float* c2b = (const float*)d_in[10];
  const float* c3w = (const float*)d_in[11]; const float* c3b = (const float*)d_in[12];
  const float* lw  = (const float*)d_in[13]; const float* lb  = (const float*)d_in[14];
  const float* wih = (const float*)d_in[15]; const float* whh = (const float*)d_in[16];
  const float* bih = (const float*)d_in[17]; const float* bhh = (const float*)d_in[18];
  const float* aw1 = (const float*)d_in[19]; const float* ab1 = (const float*)d_in[20];
  const float* aw2 = (const float*)d_in[21]; const float* ab2 = (const float*)d_in[22];
  const float* aw3 = (const float*)d_in[23]; const float* ab3 = (const float*)d_in[24];
  const float* cw1 = (const float*)d_in[25]; const float* cb1 = (const float*)d_in[26];
  const float* cw2 = (const float*)d_in[27]; const float* cb2 = (const float*)d_in[28];
  const float* cw3 = (const float*)d_in[29]; const float* cb3 = (const float*)d_in[30];
  const float* logstd = (const float*)d_in[31];

  char* ws = (char*)d_ws;
  size_t off = 0;
  auto alloc = [&](size_t bytes) -> char* {
    char* p = ws + off;
    off = (off + bytes + 255) & ~(size_t)255;
    return p;
  };

  // --- persistent (per-call) packed weights / biases ---
  h16*  w1p  = (h16*)alloc((size_t)32 * 192 * 2);
  h16*  w2p  = (h16*)alloc((size_t)64 * 512 * 2);
  h16*  w3p  = (h16*)alloc((size_t)64 * 576 * 2);
  h16*  lwp  = (h16*)alloc((size_t)512 * 3136 * 2);
  h16*  wihp = (h16*)alloc((size_t)512 * 544 * 2);
  h16*  whhp = (h16*)alloc((size_t)512 * 128 * 2);
  h16*  aw1p = (h16*)alloc((size_t)64 * 128 * 2);
  h16*  aw2p = (h16*)alloc((size_t)64 * 64 * 2);
  h16*  aw3p = (h16*)alloc((size_t)16 * 64 * 2);
  h16*  cw1p = (h16*)alloc((size_t)64 * 128 * 2);
  h16*  cw2p = (h16*)alloc((size_t)64 * 64 * 2);
  h16*  cw3p = (h16*)alloc((size_t)16 * 64 * 2);
  float* biasg = (float*)alloc(512 * 4);
  float* ab3p  = (float*)alloc(16 * 4);
  float* cb3p  = (float*)alloc(16 * 4);

  // --- activation regions (aliased: act3 over act1, post-conv bufs over act2)
  const size_t ACT1_BYTES = (size_t)2048 * 32 * 20 * 20 * 2;  // 52.4 MB
  const size_t ACT2_BYTES = (size_t)2048 * 64 * 9 * 9 * 2;    // 21.2 MB
  char* r1 = alloc(ACT1_BYTES);
  char* r2 = alloc(ACT2_BYTES);
  h16* act1 = (h16*)r1;
  h16* act2 = (h16*)r2;
  h16* act3 = (h16*)r1;                                       // act1 dead after conv2

  size_t p2 = 0;
  auto sub = [&](size_t bytes) -> char* {                     // act2 dead after conv3
    char* p = r2 + p2;
    p2 = (p2 + bytes + 255) & ~(size_t)255;
    return p;
  };
  h16*   aseq   = (h16*)  sub((size_t)2048 * 544 * 2);
  float* gatesx = (float*)sub((size_t)2048 * 512 * 4);
  float* gbuf   = (float*)sub((size_t)128 * 512 * 4);
  float* cstate = (float*)sub((size_t)128 * 128 * 4);
  h16*   hstate = (h16*)  sub((size_t)128 * 128 * 2);
  h16*   hidden = (h16*)  sub((size_t)2048 * 128 * 2);
  h16*   mu1    = (h16*)  sub((size_t)2048 * 64 * 2);
  h16*   mu2    = (h16*)  sub((size_t)2048 * 64 * 2);
  h16*   vv1    = (h16*)  sub((size_t)2048 * 64 * 2);
  h16*   vv2    = (h16*)  sub((size_t)2048 * 64 * 2);
  float* mu3    = (float*)sub((size_t)2048 * 16 * 4);
  float* vv3    = (float*)sub((size_t)2048 * 16 * 4);

  auto LP = [](int n) { return dim3((unsigned)((n + 255) / 256)); };

  // --- pack weights to f16 (with zero padding where shapes need it) ---
  k_pack_pad_f16<<<LP(32 * 192), 256, 0, stream>>>(c1w, w1p, 32, 192, 32, 192);
  k_pack_pad_f16<<<LP(64 * 512), 256, 0, stream>>>(c2w, w2p, 64, 512, 64, 512);
  k_pack_pad_f16<<<LP(64 * 576), 256, 0, stream>>>(c3w, w3p, 64, 576, 64, 576);
  k_pack_pad_f16<<<LP(512 * 3136), 256, 0, stream>>>(lw, lwp, 512, 3136, 512, 3136);
  k_pack_pad_f16<<<LP(512 * 544), 256, 0, stream>>>(wih, wihp, 512, 515, 512, 544);
  k_pack_pad_f16<<<LP(512 * 128), 256, 0, stream>>>(whh, whhp, 512, 128, 512, 128);
  k_pack_pad_f16<<<LP(64 * 128), 256, 0, stream>>>(aw1, aw1p, 64, 128, 64, 128);
  k_pack_pad_f16<<<LP(64 * 64), 256, 0, stream>>>(aw2, aw2p, 64, 64, 64, 64);
  k_pack_pad_f16<<<LP(16 * 64), 256, 0, stream>>>(aw3, aw3p, 2, 64, 16, 64);
  k_pack_pad_f16<<<LP(64 * 128), 256, 0, stream>>>(cw1, cw1p, 64, 128, 64, 128);
  k_pack_pad_f16<<<LP(64 * 64), 256, 0, stream>>>(cw2, cw2p, 64, 64, 64, 64);
  k_pack_pad_f16<<<LP(16 * 64), 256, 0, stream>>>(cw3, cw3p, 1, 64, 16, 64);
  k_add_f32<<<LP(512), 256, 0, stream>>>(bih, bhh, biasg, 512);
  k_pad_f32<<<LP(16), 256, 0, stream>>>(ab3, ab3p, 2, 16);
  k_pad_f32<<<LP(16), 256, 0, stream>>>(cb3, cb3p, 1, 16);

  // --- conv tower (implicit GEMM, WMMA; one wave covers all COUT) ---
  conv_wmma_f16<3, 8, 8, 4, 84, 84, 32, 20, 20, true>
      <<<dim3(51200u), 32, 0, stream>>>(x, w1p, c1b, act1);
  conv_wmma_f16<32, 4, 4, 2, 20, 20, 64, 9, 9, false>
      <<<dim3(10368u), 32, 0, stream>>>(act1, w2p, c2b, act2);
  conv_wmma_f16<64, 3, 3, 1, 9, 9, 64, 7, 7, false>
      <<<dim3(6272u), 32, 0, stream>>>(act2, w3p, c3b, act3);

  // --- linear 3136->512 (+relu) straight into A_seq[:, :512] (ldc = 544) ---
  gemm_wmma_f16<4, ACT_RELU, true, false, false><<<dim3(256), 128, 0, stream>>>(
      act3, 3136, lwp, 3136, lb, nullptr, 0, aseq, 544, nullptr, 2048, 512, 3136);
  k_seq_tail<<<LP(2048), 256, 0, stream>>>(la, lr, aseq);

  // --- gates_x = A_seq @ W_ih^T + (b_ih + b_hh)  (all 16 steps at once) ---
  gemm_wmma_f16<4, ACT_NONE, false, false, false><<<dim3(256), 128, 0, stream>>>(
      aseq, 544, wihp, 544, biasg, nullptr, 0, gatesx, 512, nullptr, 2048, 512, 544);

  // --- LSTM recurrence: 16 sequential (GEMM + elementwise) steps ---
  k_init_state<<<LP(128 * 128), 256, 0, stream>>>(h0, c0, hstate, cstate);
  for (int t = 0; t < 16; t++) {
    gemm_wmma_f16<4, ACT_NONE, false, true, true><<<dim3(16), 128, 0, stream>>>(
        hstate, 128, whhp, 128, nullptr,
        gatesx + (size_t)t * 128 * 512, 512,
        gbuf, 512, done + t * 128, 128, 512, 128);
    k_lstm_elem<<<LP(128 * 128), 256, 0, stream>>>(
        gbuf, done + t * 128, cstate, hstate, hidden + (size_t)t * 128 * 128);
  }

  // --- actor head ---
  gemm_wmma_f16<4, ACT_TANH, true, false, false><<<dim3(32), 128, 0, stream>>>(
      hidden, 128, aw1p, 128, ab1, nullptr, 0, mu1, 64, nullptr, 2048, 64, 128);
  gemm_wmma_f16<4, ACT_TANH, true, false, false><<<dim3(32), 128, 0, stream>>>(
      mu1, 64, aw2p, 64, ab2, nullptr, 0, mu2, 64, nullptr, 2048, 64, 64);
  gemm_wmma_f16<1, ACT_NONE, false, false, false><<<dim3(32), 128, 0, stream>>>(
      mu2, 64, aw3p, 64, ab3p, nullptr, 0, mu3, 16, nullptr, 2048, 16, 64);

  // --- critic head ---
  gemm_wmma_f16<4, ACT_TANH, true, false, false><<<dim3(32), 128, 0, stream>>>(
      hidden, 128, cw1p, 128, cb1, nullptr, 0, vv1, 64, nullptr, 2048, 64, 128);
  gemm_wmma_f16<4, ACT_TANH, true, false, false><<<dim3(32), 128, 0, stream>>>(
      vv1, 64, cw2p, 64, cb2, nullptr, 0, vv2, 64, nullptr, 2048, 64, 64);
  gemm_wmma_f16<1, ACT_NONE, false, false, false><<<dim3(32), 128, 0, stream>>>(
      vv2, 64, cw3p, 64, cb3p, nullptr, 0, vv3, 16, nullptr, 2048, 16, 64);

  // --- log-prob / entropy / value ---
  k_finalize<<<LP(2048), 256, 0, stream>>>(mu3, vv3, actn, logstd, (float*)d_out);
}
